// EntityAttentionLayer_15762529977056
// MI455X (gfx1250) — compile-verified, matrix operand order checked by reading, not run on hardware
//
#include <hip/hip_runtime.h>
#include <hip/hip_bf16.h>

typedef unsigned short u16t;
typedef __attribute__((ext_vector_type(16))) __bf16 v16bf;
typedef __attribute__((ext_vector_type(8)))  float  v8f;
typedef __attribute__((ext_vector_type(4)))  unsigned int u32x4;
typedef __attribute__((ext_vector_type(8)))  int i32x8;
typedef __attribute__((ext_vector_type(4)))  int i32x4;

#define NEGV (-100000000.0f)

// round-to-nearest-even f32 -> bf16 (raw u16)
__device__ __forceinline__ u16t f2bf(float f) {
  union { float f; unsigned u; } x; x.f = f;
  unsigned r = x.u + 0x7FFFu + ((x.u >> 16) & 1u);
  return (u16t)(r >> 16);
}

// WMMA 16-bit operand load (bf16 source).
// p = row_base + k0 + 8*(lane>>4); elements 0..7 at p[0..7], 8..15 at p[16..23].
__device__ __forceinline__ v16bf ld_ab(const u16t* p) {
  union { v16bf v; uint4 u[2]; } r;
  r.u[0] = *(const uint4*)(p);
  r.u[1] = *(const uint4*)(p + 16);
  return r.v;
}

// WMMA A operand straight from f32 memory with in-register bf16 conversion.
__device__ __forceinline__ v16bf ld_a_f32(const float* p) {
  union { v16bf v; __bf16 e[16]; } r;
#pragma unroll
  for (int e = 0; e < 8; ++e) {
    r.e[e]     = (__bf16)p[e];
    r.e[8 + e] = (__bf16)p[16 + e];
  }
  return r.v;
}

// ---------------- elementwise convert f32 -> bf16 (weights only) ----------------
__global__ void k_convert(const float* __restrict__ in, u16t* __restrict__ out, int n) {
  int i = blockIdx.x * blockDim.x + threadIdx.x;
  if (i < n) out[i] = f2bf(in[i]);
}

// ---------------- left_n per batch: ceil(0.5 * #(entity_mask==0)) ----------------
__global__ void k_leftn(const int* __restrict__ emask, int* __restrict__ leftn) {
  int b = blockIdx.x * blockDim.x + threadIdx.x;
  if (b >= 512) return;
  int cnt = 0;
  for (int i = 0; i < 128; ++i) cnt += (emask[b * 128 + i] == 0) ? 1 : 0;
  leftn[b] = (cnt + 1) >> 1;
}

// ---------------- QKV GEMM: entities(65536x512,f32) * W_in^T(512x1536,bf16) ------
// One wave per 16x64 output block. A converted f32->bf16 in registers (no staging).
__global__ void k_qkv(const float* __restrict__ X, const u16t* __restrict__ W,
                      u16t* __restrict__ qbf, u16t* __restrict__ kbf,
                      u16t* __restrict__ vTbf) {
  const int tm  = blockIdx.x;             // 0..4095 (16 rows each)
  const int tn4 = blockIdx.y;             // 0..23   (64 cols each)
  if (tn4 < 8 && (tm & 7) != 0) return;   // q only for first 16 of every 128 rows
  const int lane = threadIdx.x;
  const int lh = lane >> 4, ll = lane & 15;
  const float* arow = X + (size_t)(tm * 16 + ll) * 512 + 8 * lh;
  const u16t*  brow = W + (size_t)(tn4 * 64 + ll) * 512 + 8 * lh;
  v8f acc[4] = {};
#pragma unroll
  for (int k0 = 0; k0 < 512; k0 += 32) {
    v16bf a = ld_a_f32(arow + k0);
#pragma unroll
    for (int t = 0; t < 4; ++t) {
      v16bf b = ld_ab(brow + (size_t)t * 16 * 512 + k0);
      acc[t] = __builtin_amdgcn_wmma_f32_16x16x32_bf16(false, a, false, b,
                                                       (short)0, acc[t], false, false);
    }
  }
#pragma unroll
  for (int t = 0; t < 4; ++t) {
    const int n_global = tn4 * 64 + t * 16 + ll;
#pragma unroll
    for (int r = 0; r < 8; ++r) {
      int m_global = tm * 16 + r + 8 * lh;
      int b = m_global >> 7;        // batch
      int ent = m_global & 127;     // entity index
      u16t val = f2bf(acc[t][r]);
      if (n_global < 512) {         // q
        if (ent < 16) qbf[(size_t)(b * 16 + ent) * 512 + n_global] = val;
      } else if (n_global < 1024) { // k
        kbf[(size_t)(b * 128 + ent) * 512 + (n_global - 512)] = val;
      } else {                      // v, stored transposed per head: (b,h,hd,ne)
        int col = n_global - 1024;
        int h = col >> 6, d = col & 63;
        vTbf[(size_t)((b * 8 + h) * 64 + d) * 128 + ent] = val;
      }
    }
  }
}

// ---------------- fused attention per (b,h): logits + softmax + P*V ----------
// One wave per (b,h). 16x128 logits in 8 WMMA accumulators. V panel (16KB,
// contiguous) pulled into LDS by the Tensor Data Mover while logits/softmax run.
__global__ void k_attn(const u16t* __restrict__ qbf, const u16t* __restrict__ kbf,
                       const u16t* __restrict__ vTbf, float* __restrict__ logits,
                       u16t* __restrict__ obf) {
  __shared__ u16t smem_p[16 * 128];       // P tile, row-major bf16
  __shared__ u16t smem_v[64 * 128];       // V panel (hd x ne), bf16
  const int bh = blockIdx.x;              // 0..4095
  const int b = bh >> 3, h = bh & 7;
  const int lane = threadIdx.x;
  const int lh = lane >> 4, ll = lane & 15;

#if __has_builtin(__builtin_amdgcn_tensor_load_to_lds)
  // ---- TDM: async-load this (b,h)'s V panel (64x128 bf16 = 16KB, contiguous) ----
  {
    unsigned long long ga = (unsigned long long)(const void*)(vTbf + (size_t)bh * 8192);
    unsigned lds_off = (unsigned)(size_t)(void*)smem_v;
    u32x4 g0;
    g0[0] = 1u;                                     // count=1 (valid user descriptor)
    g0[1] = lds_off;                                // lds_addr
    g0[2] = (unsigned)(ga & 0xFFFFFFFFu);           // global_addr[31:0]
    g0[3] = (unsigned)((ga >> 32) & 0x1FFFFFFu) | (2u << 30); // global_addr[56:32], type=2
    i32x8 g1;
    g1[0] = (int)(1u << 16);          // data_size = 2B
    g1[1] = (int)(8192u << 16);       // tensor_dim0 = 8192 (bits 79:48, low half)
    g1[2] = (int)(1u << 16);          // tensor_dim1 = 1   (bits 111:80, low half)
    g1[3] = (int)(8192u << 16);       // tile_dim0 = 8192  (bits 127:112)
    g1[4] = 1;                        // tile_dim1 = 1
    g1[5] = 8192;                     // tensor_dim0_stride = 8192
    g1[6] = 0;
    g1[7] = 0;
    i32x4 gz4 = {0, 0, 0, 0};
    i32x8 gz8 = {0, 0, 0, 0, 0, 0, 0, 0};
    // clang-23 / therock-10.0 arity: (g0, g1, g2, g3, g4, cpol)
    __builtin_amdgcn_tensor_load_to_lds(g0, g1, gz4, gz4, gz8, 0);
  }
#endif

  // ---- logits: q(16x64) . k^T(64x128), 8 column tiles ----
  const u16t* arow = qbf + (size_t)(b * 16 + ll) * 512 + h * 64 + 8 * lh;
  const u16t* brow = kbf + (size_t)(b * 128 + ll) * 512 + h * 64 + 8 * lh;
  v8f acc[8] = {};
#pragma unroll
  for (int k0 = 0; k0 < 64; k0 += 32) {
    v16bf a = ld_ab(arow + k0);
#pragma unroll
    for (int t = 0; t < 8; ++t) {
      v16bf bb = ld_ab(brow + (size_t)t * 16 * 512 + k0);
      acc[t] = __builtin_amdgcn_wmma_f32_16x16x32_bf16(false, a, false, bb,
                                                       (short)0, acc[t], false, false);
    }
  }
  // scale by 1/sqrt(64) and spill f32 logits for the rank kernel
#pragma unroll
  for (int t = 0; t < 8; ++t)
#pragma unroll
    for (int r = 0; r < 8; ++r) {
      acc[t][r] *= 0.125f;
      int m = r + 8 * lh;
      logits[(size_t)(bh * 16 + m) * 128 + t * 16 + ll] = acc[t][r];
    }

  // ---- row softmax (row m = r + 8*lh lives in one 16-lane half) ----
  float inv[8];
#pragma unroll
  for (int r = 0; r < 8; ++r) {
    float mx = acc[0][r];
#pragma unroll
    for (int t = 1; t < 8; ++t) mx = fmaxf(mx, acc[t][r]);
    for (int off = 8; off > 0; off >>= 1) mx = fmaxf(mx, __shfl_xor(mx, off, 32));
    float s = 0.0f;
#pragma unroll
    for (int t = 0; t < 8; ++t) {
      float e = __expf(acc[t][r] - mx);
      acc[t][r] = e;
      s += e;
    }
    for (int off = 8; off > 0; off >>= 1) s += __shfl_xor(s, off, 32);
    inv[r] = 1.0f / s;
  }

  // ---- P (bf16) -> LDS in row-major, to re-read in WMMA A layout ----
#pragma unroll
  for (int t = 0; t < 8; ++t)
#pragma unroll
    for (int r = 0; r < 8; ++r) {
      int m = r + 8 * lh;
      smem_p[m * 128 + t * 16 + ll] = f2bf(acc[t][r] * inv[r]);
    }
  __syncthreads();
#if __has_builtin(__builtin_amdgcn_tensor_load_to_lds)
  __builtin_amdgcn_s_wait_tensorcnt(0);   // V panel resident in LDS
#endif

  // ---- O = P(16x128) . V(128x64), 4 d-tiles; both operands from LDS ----
  v8f oacc[4] = {};
#pragma unroll
  for (int k0 = 0; k0 < 128; k0 += 32) {
    v16bf a = ld_ab(smem_p + ll * 128 + k0 + 8 * lh);
#pragma unroll
    for (int dt = 0; dt < 4; ++dt) {
#if __has_builtin(__builtin_amdgcn_tensor_load_to_lds)
      v16bf bb = ld_ab(smem_v + (dt * 16 + ll) * 128 + k0 + 8 * lh);
#else
      v16bf bb = ld_ab(vTbf + (size_t)(bh * 64 + dt * 16 + ll) * 128 + k0 + 8 * lh);
#endif
      oacc[dt] = __builtin_amdgcn_wmma_f32_16x16x32_bf16(false, a, false, bb,
                                                         (short)0, oacc[dt], false, false);
    }
  }
#pragma unroll
  for (int dt = 0; dt < 4; ++dt)
#pragma unroll
    for (int r = 0; r < 8; ++r) {
      int m = r + 8 * lh;        // query index
      obf[(size_t)(b * 16 + m) * 512 + h * 64 + dt * 16 + ll] = f2bf(oacc[dt][r]);
    }
}

// ---------------- rank kernel -> true_pre_mask (stable-ascending argsort rank) ----
__global__ void k_rank(const float* __restrict__ logits, const int* __restrict__ pre_mask,
                       const int* __restrict__ leftn, float* __restrict__ out_tpm) {
  __shared__ float sm[128];
  int row = blockIdx.x;          // bh*16 + q ; bh = b*8 + h
  int n = threadIdx.x;           // 0..127
  int b = row >> 7;
  int q = row & 15;
  int pre = pre_mask[(size_t)(b * 16 + q) * 128 + n];
  float my = (pre > 0) ? NEGV : logits[(size_t)row * 128 + n];
  sm[n] = my;
  __syncthreads();
  int rank = 0;
  for (int j = 0; j < 128; ++j) {
    float v = sm[j];
    rank += (v < my || (v == my && j < n)) ? 1 : 0;
  }
  float o;
  if (pre > 0) o = 1.0f;
  else o = (rank >= 128 - leftn[b]) ? 0.0f : 1.0f;
  out_tpm[(size_t)row * 128 + n] = o;
}

// ---------------- output GEMM: O(8192x512)*W_out^T + b_out, post-mask ----------
// One wave per 16x64 output block.
__global__ void k_out(const u16t* __restrict__ obf, const u16t* __restrict__ wbf,
                      const float* __restrict__ b_out, const int* __restrict__ post_mask,
                      float* __restrict__ out) {
  int tm  = blockIdx.x;          // 0..511
  int tn4 = blockIdx.y;          // 0..7
  int lane = threadIdx.x, lh = lane >> 4, ll = lane & 15;
  const u16t* arow = obf + (size_t)(tm * 16 + ll) * 512 + 8 * lh;
  const u16t* brow = wbf + (size_t)(tn4 * 64 + ll) * 512 + 8 * lh;
  v8f acc[4] = {};
#pragma unroll
  for (int k0 = 0; k0 < 512; k0 += 32) {
    v16bf a = ld_ab(arow + k0);
#pragma unroll
    for (int t = 0; t < 4; ++t) {
      v16bf bb = ld_ab(brow + (size_t)t * 16 * 512 + k0);
      acc[t] = __builtin_amdgcn_wmma_f32_16x16x32_bf16(false, a, false, bb,
                                                       (short)0, acc[t], false, false);
    }
  }
#pragma unroll
  for (int t = 0; t < 4; ++t) {
    int n = tn4 * 64 + t * 16 + ll;
    float bias = b_out[n];
#pragma unroll
    for (int r = 0; r < 8; ++r) {
      int m = tm * 16 + r + 8 * lh;   // = b*16 + q
      float y = acc[t][r] + bias;
      if (post_mask[m] > 0) y = 0.0f;
      out[(size_t)m * 512 + n] = y;
    }
  }
}

extern "C" void kernel_launch(void* const* d_in, const int* in_sizes, int n_in,
                              void* d_out, int out_size, void* d_ws, size_t ws_size,
                              hipStream_t stream) {
  const float* entities = (const float*)d_in[0];  // (512,128,512)
  const float* W_in     = (const float*)d_in[1];  // (1536,512)
  const float* W_out    = (const float*)d_in[2];  // (512,512)
  const float* b_out    = (const float*)d_in[3];  // (512,)
  const int*   pre_mask = (const int*)d_in[4];    // (512,16,128)
  const int*   post_mask= (const int*)d_in[5];    // (512,16)
  const int*   ent_mask = (const int*)d_in[6];    // (512,128)
  float* out = (float*)d_out;                     // [4194304 attn | 8388608 tpm]

  // workspace carve-up (256B aligned)
  char* w = (char*)d_ws;
  size_t off = 0;
  auto carve = [&](size_t bytes) -> char* {
    char* p = w + off;
    off += (bytes + 255) & ~(size_t)255;
    return p;
  };
  u16t*  wibf   = (u16t*)carve((size_t)1536 * 512 * 2);    // W_in bf16
  u16t*  wobf   = (u16t*)carve((size_t)512 * 512 * 2);     // W_out bf16
  u16t*  qbf    = (u16t*)carve((size_t)512 * 16 * 512 * 2);
  u16t*  kbf    = (u16t*)carve((size_t)512 * 128 * 512 * 2);
  u16t*  vtbf   = (u16t*)carve((size_t)512 * 128 * 512 * 2);
  float* logits = (float*)carve((size_t)65536 * 128 * 4);
  u16t*  obf    = (u16t*)carve((size_t)8192 * 512 * 2);
  int*   leftn  = (int*)carve((size_t)512 * 4);
  if (off > ws_size) return;  // workspace too small; do nothing (deterministic)

  // 1) convert weights to bf16 (entities converted in-register inside k_qkv)
  k_convert<<<(786432 + 255) / 256, 256, 0, stream>>>(W_in, wibf, 786432);
  k_convert<<<(262144 + 255) / 256, 256, 0, stream>>>(W_out, wobf, 262144);

  // 2) per-batch left_n
  k_leftn<<<2, 256, 0, stream>>>(ent_mask, leftn);

  // 3) QKV GEMM (WMMA bf16, 16x64 blocks, fused f32->bf16 A conversion)
  k_qkv<<<dim3(4096, 24), 32, 0, stream>>>(entities, wibf, qbf, kbf, vtbf);

  // 4) fused logits + softmax + P*V (WMMA bf16, TDM V staging, LDS transpose)
  k_attn<<<4096, 32, 0, stream>>>(qbf, kbf, vtbf, logits, obf);

  // 5) rank / true_pre_mask (second output)
  k_rank<<<65536, 128, 0, stream>>>(logits, pre_mask, leftn, out + 4194304);

  // 6) output projection + bias + post-mask (WMMA bf16, 16x64 blocks)
  k_out<<<dim3(512, 8), 32, 0, stream>>>(obf, wobf, b_out, post_mask, out);
}
